// MSAAttention_89850715833159
// MI455X (gfx1250) — matrix operand order repaired and checked by reading, not compiled, not run on hardware
//
#include <hip/hip_runtime.h>
#include <hip/hip_bf16.h>

// ---------------- types / helpers ----------------
typedef __attribute__((ext_vector_type(16))) __bf16          v16bf;
typedef __attribute__((ext_vector_type(16))) unsigned short  v16us;
typedef __attribute__((ext_vector_type(8)))  float           v8f;

// match the async-builtin parameter types: GCC-style int vectors
typedef int gvec2i __attribute__((vector_size(8)));
typedef int gvec4i __attribute__((vector_size(16)));

#define AS_GLOBAL __attribute__((address_space(1)))
#define AS_LDS    __attribute__((address_space(3)))

#if __has_builtin(__builtin_amdgcn_global_load_async_to_lds_b128)
#define HAVE_ASYNC128 1
#endif
#if __has_builtin(__builtin_amdgcn_global_load_async_to_lds_b64)
#define HAVE_ASYNC64 1
#endif

__device__ __forceinline__ void async_load_b128(const void* g, void* l) {
#if defined(HAVE_ASYNC128)
  __builtin_amdgcn_global_load_async_to_lds_b128(
      (AS_GLOBAL gvec4i*)g, (AS_LDS gvec4i*)l, 0, 0);
#endif
}
__device__ __forceinline__ void async_load_b64(const void* g, void* l) {
#if defined(HAVE_ASYNC64)
  __builtin_amdgcn_global_load_async_to_lds_b64(
      (AS_GLOBAL gvec2i*)g, (AS_LDS gvec2i*)l, 0, 0);
#endif
}

__device__ __forceinline__ void wait_async0() {
#if __has_builtin(__builtin_amdgcn_s_wait_asynccnt)
  __builtin_amdgcn_s_wait_asynccnt(0);
#else
  asm volatile("s_wait_asynccnt 0x0" ::: "memory");
#endif
}

#if __has_builtin(__builtin_amdgcn_exp2f)
#define EXP2F(x) __builtin_amdgcn_exp2f(x)
#else
#define EXP2F(x) exp2f(x)
#endif
#if __has_builtin(__builtin_amdgcn_rcpf)
#define RCPF(x) __builtin_amdgcn_rcpf(x)
#else
#define RCPF(x) (1.0f / (x))
#endif

#define LOG2E 1.4426950408889634f

__device__ __forceinline__ unsigned short f2bf(float f) {
  unsigned u = __builtin_bit_cast(unsigned, f);
  u += 0x7FFFu + ((u >> 16) & 1u);            // round-to-nearest-even
  return (unsigned short)(u >> 16);
}
__device__ __forceinline__ float bf2f(unsigned short h) {
  unsigned u = ((unsigned)h) << 16;
  return __builtin_bit_cast(float, u);
}
__device__ __forceinline__ float wave_sum(float v) {
  v += __shfl_xor(v, 1, 32);
  v += __shfl_xor(v, 2, 32);
  v += __shfl_xor(v, 4, 32);
  v += __shfl_xor(v, 8, 32);
  v += __shfl_xor(v, 16, 32);
  return v;
}

// A-fragment K index for 16-bit WMMA 16x32: contiguous order, element i -> K = i + 16*half
// (matches the B-fragment order).
__device__ __forceinline__ int kidxA(int i, int half) {
  return i + 16 * half;
}
// B-fragment (32x16): lane = N (mod 16), element i -> K = i + 16*(lane>=16)

__device__ __forceinline__ v8f wmma_bf16(v16us a, v16us b, v8f c) {
  return __builtin_amdgcn_wmma_f32_16x16x32_bf16(
      false, __builtin_bit_cast(v16bf, a),
      false, __builtin_bit_cast(v16bf, b),
      (short)0, c, false, false);
}

#define S_DIM   256
#define R_DIM   256
#define CIN     256
#define CZ      128
#define HH      8
#define CH      32
#define NQKVG   1024

// ---------------- 1) LayerNorm on m -> bf16 (one wave per row, no barriers) ----
__global__ __launch_bounds__(256) void k_ln_m(const float* __restrict__ m,
                                              const float* __restrict__ g,
                                              const float* __restrict__ b,
                                              unsigned short* __restrict__ mln) {
  const int t = threadIdx.x, wave = t >> 5, lane = t & 31;
  const int row = blockIdx.x * 8 + wave;
  const float4* src = (const float4*)(m + (size_t)row * CIN);
  float4 x0 = src[lane], x1 = src[lane + 32];
  float sm = x0.x + x0.y + x0.z + x0.w + x1.x + x1.y + x1.z + x1.w;
  float mu = wave_sum(sm) * (1.0f / CIN);
  float d0 = x0.x - mu, d1 = x0.y - mu, d2 = x0.z - mu, d3 = x0.w - mu;
  float d4 = x1.x - mu, d5 = x1.y - mu, d6 = x1.z - mu, d7 = x1.w - mu;
  float vs = d0*d0 + d1*d1 + d2*d2 + d3*d3 + d4*d4 + d5*d5 + d6*d6 + d7*d7;
  float rstd = rsqrtf(wave_sum(vs) * (1.0f / CIN) + 1e-5f);
  float4 ga = ((const float4*)g)[lane], gb = ((const float4*)g)[lane + 32];
  float4 ba = ((const float4*)b)[lane], bb = ((const float4*)b)[lane + 32];
  ushort4 o0, o1;
  o0.x = f2bf(d0 * rstd * ga.x + ba.x);
  o0.y = f2bf(d1 * rstd * ga.y + ba.y);
  o0.z = f2bf(d2 * rstd * ga.z + ba.z);
  o0.w = f2bf(d3 * rstd * ga.w + ba.w);
  o1.x = f2bf(d4 * rstd * gb.x + bb.x);
  o1.y = f2bf(d5 * rstd * gb.y + bb.y);
  o1.z = f2bf(d6 * rstd * gb.z + bb.z);
  o1.w = f2bf(d7 * rstd * gb.w + bb.w);
  *(ushort4*)(mln + (size_t)row * CIN + lane * 4)       = o0;
  *(ushort4*)(mln + (size_t)row * CIN + 128 + lane * 4) = o1;
}

// ---------------- 2) pair bias (one wave per (i,j), no barriers) ----------------
__global__ __launch_bounds__(256) void k_zbias(const float* __restrict__ z,
                                               const float* __restrict__ g,
                                               const float* __restrict__ b,
                                               const float* __restrict__ w_z,
                                               float* __restrict__ zb) {
  const int t = threadIdx.x, wave = t >> 5, lane = t & 31;
  const int ij = blockIdx.x * 8 + wave;
  float4 x = ((const float4*)(z + (size_t)ij * CZ))[lane];
  float mu = wave_sum(x.x + x.y + x.z + x.w) * (1.0f / CZ);
  float d0 = x.x - mu, d1 = x.y - mu, d2 = x.z - mu, d3 = x.w - mu;
  float rstd = rsqrtf(wave_sum(d0*d0 + d1*d1 + d2*d2 + d3*d3) * (1.0f / CZ) + 1e-5f);
  float4 gg = ((const float4*)g)[lane], bb = ((const float4*)b)[lane];
  float y0 = d0 * rstd * gg.x + bb.x;
  float y1 = d1 * rstd * gg.y + bb.y;
  float y2 = d2 * rstd * gg.z + bb.z;
  float y3 = d3 * rstd * gg.w + bb.w;
  // w_z rows lane*4 .. lane*4+3, each row = 8 floats = 2 float4
  const float4* wz4 = (const float4*)w_z;
  float4 wa0 = wz4[(lane*4 + 0)*2], wb0 = wz4[(lane*4 + 0)*2 + 1];
  float4 wa1 = wz4[(lane*4 + 1)*2], wb1 = wz4[(lane*4 + 1)*2 + 1];
  float4 wa2 = wz4[(lane*4 + 2)*2], wb2 = wz4[(lane*4 + 2)*2 + 1];
  float4 wa3 = wz4[(lane*4 + 3)*2], wb3 = wz4[(lane*4 + 3)*2 + 1];
  float p[8];
  p[0] = y0*wa0.x + y1*wa1.x + y2*wa2.x + y3*wa3.x;
  p[1] = y0*wa0.y + y1*wa1.y + y2*wa2.y + y3*wa3.y;
  p[2] = y0*wa0.z + y1*wa1.z + y2*wa2.z + y3*wa3.z;
  p[3] = y0*wa0.w + y1*wa1.w + y2*wa2.w + y3*wa3.w;
  p[4] = y0*wb0.x + y1*wb1.x + y2*wb2.x + y3*wb3.x;
  p[5] = y0*wb0.y + y1*wb1.y + y2*wb2.y + y3*wb3.y;
  p[6] = y0*wb0.z + y1*wb1.z + y2*wb2.z + y3*wb3.z;
  p[7] = y0*wb0.w + y1*wb1.w + y2*wb2.w + y3*wb3.w;
  #pragma unroll
  for (int h = 0; h < HH; ++h) p[h] = wave_sum(p[h]);
  if (lane == 0) {
    #pragma unroll
    for (int h = 0; h < HH; ++h)
      zb[(size_t)h * (R_DIM * R_DIM) + ij] = LOG2E * p[h];
  }
}

// ---------------- 3) weights -> bf16, TRANSPOSED: wcT[n][k], woT[n][k] ----------------
__global__ __launch_bounds__(256) void k_cvt_w(const float* __restrict__ wq,
                                               const float* __restrict__ wk,
                                               const float* __restrict__ wv,
                                               const float* __restrict__ wg,
                                               const float* __restrict__ wo,
                                               unsigned short* __restrict__ wcT,
                                               unsigned short* __restrict__ woT) {
  int idx = blockIdx.x * 256 + threadIdx.x;   // 0..327679
  if (idx < CIN * NQKVG) {
    int n = idx >> 8, k = idx & 255;          // wcT[n*256+k] = W[k][n]
    const float* src = (n < 256) ? wq : (n < 512) ? wk : (n < 768) ? wv : wg;
    wcT[idx] = f2bf(src[k * 256 + (n & 255)]);
  } else {
    int r = idx - CIN * NQKVG;                // woT[n*256+k] = wo[k][n]
    int n = r >> 8, k = r & 255;
    woT[r] = f2bf(wo[k * 256 + n]);
  }
}

// ---------------- 4) fused QKVG projection (double-buffered async) ----------------
__global__ __launch_bounds__(256) void k_proj(const unsigned short* __restrict__ mln,
                                              const unsigned short* __restrict__ wcT,
                                              const float* __restrict__ b_g,
                                              unsigned short* __restrict__ qkvg,
                                              unsigned short* __restrict__ vT) {
  __shared__ __align__(16) unsigned short As[2][32 * 32];    // [m][k]
  __shared__ __align__(16) unsigned short Wt[2][256 * 32];   // [n][k]
  const int mbase = blockIdx.x * 32;
  const int nbase = blockIdx.y * 256;
  const int t = threadIdx.x, wave = t >> 5, lane = t & 31;
  const int Mi = wave >> 2, nq = wave & 3;
  const int mrow = lane & 15, half = lane >> 4;

  auto stage = [&](int kk, int buf) {
    const unsigned short* aG = mln + (size_t)(mbase + (t >> 3)) * CIN + kk + (t & 7) * 4;
    const unsigned short* wG = wcT + (size_t)(nbase + t) * 256 + kk;
#if defined(HAVE_ASYNC128) && defined(HAVE_ASYNC64)
    async_load_b64(aG, &As[buf][t * 4]);
    #pragma unroll
    for (int i = 0; i < 4; ++i)
      async_load_b128(wG + i * 8, &Wt[buf][t * 32 + i * 8]);
#else
    *(uint2*)(&As[buf][t * 4]) = *(const uint2*)aG;
    #pragma unroll
    for (int i = 0; i < 4; ++i)
      ((uint4*)(&Wt[buf][t * 32]))[i] = ((const uint4*)wG)[i];
#endif
  };

  v8f acc[4];
  #pragma unroll
  for (int jj = 0; jj < 4; ++jj) acc[jj] = v8f{0,0,0,0,0,0,0,0};

  stage(0, 0);
  wait_async0();
  __syncthreads();
  int cur = 0;
  for (int k0 = 0; k0 < CIN; k0 += 32) {
    int nxt = cur ^ 1;
    if (k0 + 32 < CIN) stage(k0 + 32, nxt);       // prefetch next tile
    v16us a, bfr[4];
    #pragma unroll
    for (int i = 0; i < 16; ++i) a[i] = As[cur][(Mi * 16 + mrow) * 32 + kidxA(i, half)];
    #pragma unroll
    for (int jj = 0; jj < 4; ++jj) {
      int ncol = (nq * 4 + jj) * 16 + (lane & 15);
      #pragma unroll
      for (int i = 0; i < 16; ++i) bfr[jj][i] = Wt[cur][ncol * 32 + i + 16 * half];
    }
    #pragma unroll
    for (int jj = 0; jj < 4; ++jj) acc[jj] = wmma_bf16(a, bfr[jj], acc[jj]);
    wait_async0();
    __syncthreads();
    cur = nxt;
  }

  if (nbase == 512) {
    // V block: emit transposed to vT[((s*8+h)*32+ch)*256 + key], b128 stores
    #pragma unroll
    for (int jj = 0; jj < 4; ++jj) {
      int hcol = (nq * 4 + jj) * 16 + (lane & 15);
      int row0 = mbase + Mi * 16 + 8 * half;
      int hh = hcol >> 5, ch = hcol & 31;
      int sIdx = row0 >> 8, key0 = row0 & 255;
      unsigned p0 = (unsigned)f2bf(acc[jj][0]) | ((unsigned)f2bf(acc[jj][1]) << 16);
      unsigned p1 = (unsigned)f2bf(acc[jj][2]) | ((unsigned)f2bf(acc[jj][3]) << 16);
      unsigned p2 = (unsigned)f2bf(acc[jj][4]) | ((unsigned)f2bf(acc[jj][5]) << 16);
      unsigned p3 = (unsigned)f2bf(acc[jj][6]) | ((unsigned)f2bf(acc[jj][7]) << 16);
      uint4* dst = (uint4*)(vT + ((size_t)(sIdx * HH + hh) * CH + ch) * R_DIM + key0);
      *dst = make_uint4(p0, p1, p2, p3);
    }
  } else {
    // q/k/g: uniform branch on nbase (no per-lane divergence), bounce via LDS
    unsigned short* Eb = &Wt[0][0];               // 16 KB overlay, tiles are dead
    const float qscale = 0.17677669529663687f * LOG2E;   // log2e / sqrt(32)
    #pragma unroll
    for (int jj = 0; jj < 4; ++jj) {
      int ncol = (nq * 4 + jj) * 16 + (lane & 15);
      #pragma unroll
      for (int r = 0; r < 8; ++r) {
        float v = acc[jj][r];
        if (nbase == 0)        v *= qscale;                                   // Q
        else if (nbase == 768) v = RCPF(1.0f + EXP2F(-LOG2E * (v + b_g[ncol])));
        Eb[(Mi * 16 + 8 * half + r) * 256 + ncol] = f2bf(v);
      }
    }
    __syncthreads();
    const uint4* srcv = (const uint4*)(Eb + (t >> 3) * 256 + (t & 7) * 32);
    uint4* dstv = (uint4*)(qkvg + (size_t)(mbase + (t >> 3)) * NQKVG + nbase + (t & 7) * 32);
    #pragma unroll
    for (int i = 0; i < 4; ++i) dstv[i] = srcv[i];
  }
}

// ---------------- 5) fused attention per (s,h) ----------------
__global__ __launch_bounds__(256) void k_attn(const unsigned short* __restrict__ qkvg,
                                              const unsigned short* __restrict__ vT,
                                              const float* __restrict__ mask,
                                              const float* __restrict__ zb,
                                              unsigned short* __restrict__ og) {
  __shared__ __align__(16) unsigned short Qs[R_DIM * CH];   // [q][c]
  __shared__ __align__(16) unsigned short Ks[R_DIM * CH];   // [k][c]
  __shared__ __align__(16) unsigned short Vt[CH * R_DIM];   // [c][k]
  __shared__ __align__(16) unsigned short Ps[8 * 512];      // per-wave bounce
  __shared__ __align__(16) unsigned short Gb[8 * 512];      // per-wave gate tile
  const int s = blockIdx.x >> 3, h = blockIdx.x & 7;
  const int t = threadIdx.x, wave = t >> 5, lane = t & 31;
  const int mrow = lane & 15, half = lane >> 4;

  { // stage Q,K (row-major) and V (contiguous transposed tile)
    const unsigned short* gq = qkvg + (size_t)s * (R_DIM * NQKVG) + h * CH;
    const unsigned short* gv = vT + (size_t)(s * HH + h) * (CH * R_DIM);
#if defined(HAVE_ASYNC128)
    #pragma unroll
    for (int i = 0; i < 4; ++i) {
      async_load_b128(gq + (size_t)t * NQKVG + i * 8,       Qs + t * CH + i * 8);
      async_load_b128(gq + (size_t)t * NQKVG + 256 + i * 8, Ks + t * CH + i * 8);
      async_load_b128(gv + (size_t)(i * 256 + t) * 8,       Vt + (i * 256 + t) * 8);
    }
    wait_async0();
#else
    #pragma unroll
    for (int i = 0; i < 4; ++i) {
      ((uint4*)Qs)[t * 4 + i] = ((const uint4*)gq)[(size_t)t * 128 + i];
      ((uint4*)Ks)[t * 4 + i] = ((const uint4*)gq)[(size_t)t * 128 + 32 + i];
      ((uint4*)Vt)[i * 256 + t] = ((const uint4*)gv)[i * 256 + t];
    }
#endif
  }
  // mask bias depends only on lane/kt: hoist
  float mbias[16];
  #pragma unroll
  for (int kt = 0; kt < 16; ++kt)
    mbias[kt] = (LOG2E * 1e9f) * (mask[s * R_DIM + kt * 16 + (lane & 15)] - 1.0f);
  const float* zrow = zb + (size_t)h * (R_DIM * R_DIM);
  __syncthreads();

  for (int qi = 0; qi < 2; ++qi) {
    const int qt = wave + qi * 8;                 // 16-row Q tile
    unsigned short* P = &Ps[wave * 512];
    unsigned short* G = &Gb[wave * 512];
    { // prefetch gate tile [16 q][32 ch] for this q-tile into LDS (b128 path)
      const uint4* gg = (const uint4*)(qkvg + (size_t)(s * R_DIM + qt * 16 + (lane >> 1)) * NQKVG
                                       + 768 + h * CH + (lane & 1) * 16);
      uint4* gd = (uint4*)(G + (lane >> 1) * 32 + (lane & 1) * 16);
      gd[0] = gg[0]; gd[1] = gg[1];
    }
    v16us a;
    #pragma unroll
    for (int i = 0; i < 16; ++i) a[i] = Qs[(qt * 16 + mrow) * CH + kidxA(i, half)];

    // scores (log2 domain): C operand carries mask + pair bias
    v8f D[16];
    #pragma unroll
    for (int kt = 0; kt < 16; ++kt) {
      int col = kt * 16 + (lane & 15);
      v8f c;
      #pragma unroll
      for (int r = 0; r < 8; ++r)
        c[r] = mbias[kt] + zrow[(qt * 16 + 8 * half + r) * R_DIM + col];
      v16us bf;
      #pragma unroll
      for (int i = 0; i < 16; ++i) bf[i] = Ks[(kt * 16 + mrow) * CH + (i + 16 * half)];
      D[kt] = wmma_bf16(a, bf, c);
    }
    // softmax (exp2): per-lane over 16 frags, xor-shuffle across 16 lanes
    float inv[8];
    #pragma unroll
    for (int r = 0; r < 8; ++r) {
      float mx = -1e30f;
      #pragma unroll
      for (int kt = 0; kt < 16; ++kt) mx = fmaxf(mx, D[kt][r]);
      mx = fmaxf(mx, __shfl_xor(mx, 1, 32));
      mx = fmaxf(mx, __shfl_xor(mx, 2, 32));
      mx = fmaxf(mx, __shfl_xor(mx, 4, 32));
      mx = fmaxf(mx, __shfl_xor(mx, 8, 32));
      float sm = 0.0f;
      #pragma unroll
      for (int kt = 0; kt < 16; ++kt) { D[kt][r] = EXP2F(D[kt][r] - mx); sm += D[kt][r]; }
      sm += __shfl_xor(sm, 1, 32);
      sm += __shfl_xor(sm, 2, 32);
      sm += __shfl_xor(sm, 4, 32);
      sm += __shfl_xor(sm, 8, 32);
      inv[r] = RCPF(sm);
    }
    // O = P @ V via per-wave LDS repack (DS in-order within a wave)
    v8f O0 = v8f{0,0,0,0,0,0,0,0}, O1 = v8f{0,0,0,0,0,0,0,0};
    for (int kc = 0; kc < 8; ++kc) {              // 32 keys per chunk
      #pragma unroll
      for (int tt = 0; tt < 2; ++tt) {
        int kt = kc * 2 + tt;
        #pragma unroll
        for (int r = 0; r < 8; ++r)
          P[(8 * half + r) * 32 + tt * 16 + (lane & 15)] = f2bf(D[kt][r] * inv[r]);
      }
      v16us pa, b0, b1;
      #pragma unroll
      for (int i = 0; i < 16; ++i) pa[i] = P[mrow * 32 + kidxA(i, half)];
      #pragma unroll
      for (int i = 0; i < 16; ++i) {
        b0[i] = Vt[(lane & 15) * R_DIM + kc * 32 + i + 16 * half];
        b1[i] = Vt[(16 + (lane & 15)) * R_DIM + kc * 32 + i + 16 * half];
      }
      O0 = wmma_bf16(pa, b0, O0);
      O1 = wmma_bf16(pa, b1, O1);
    }
    // gate (from LDS tile) -> repack via P -> vector store
    #pragma unroll
    for (int part = 0; part < 2; ++part) {
      #pragma unroll
      for (int r = 0; r < 8; ++r) {
        float gv = bf2f(G[(8 * half + r) * 32 + part * 16 + (lane & 15)]);
        float ov = (part ? O1[r] : O0[r]) * gv;
        P[(8 * half + r) * 32 + part * 16 + (lane & 15)] = f2bf(ov);
      }
    }
    {
      const uint4* ps = (const uint4*)(P + (lane >> 1) * 32 + (lane & 1) * 16);
      uint4* dstv = (uint4*)(og + (size_t)(s * R_DIM + qt * 16 + (lane >> 1)) * (HH * CH)
                             + h * CH + (lane & 1) * 16);
      dstv[0] = ps[0]; dstv[1] = ps[1];
    }
  }
}

// ---------------- 6) output projection (double-buffered async) ----------------
__global__ __launch_bounds__(256) void k_out(const unsigned short* __restrict__ og,
                                             const unsigned short* __restrict__ woT,
                                             const float* __restrict__ b_o,
                                             float* __restrict__ out) {
  __shared__ __align__(16) unsigned short As[2][32 * 32];
  __shared__ __align__(16) unsigned short Wt[2][256 * 32];   // [n][k]
  const int mbase = blockIdx.x * 32;
  const int t = threadIdx.x, wave = t >> 5, lane = t & 31;
  const int Mi = wave >> 2, nq = wave & 3;
  const int mrow = lane & 15, half = lane >> 4;

  auto stage = [&](int kk, int buf) {
    const unsigned short* aG = og + (size_t)(mbase + (t >> 3)) * 256 + kk + (t & 7) * 4;
    const unsigned short* wG = woT + (size_t)t * 256 + kk;
#if defined(HAVE_ASYNC128) && defined(HAVE_ASYNC64)
    async_load_b64(aG, &As[buf][t * 4]);
    #pragma unroll
    for (int i = 0; i < 4; ++i)
      async_load_b128(wG + i * 8, &Wt[buf][t * 32 + i * 8]);
#else
    *(uint2*)(&As[buf][t * 4]) = *(const uint2*)aG;
    #pragma unroll
    for (int i = 0; i < 4; ++i)
      ((uint4*)(&Wt[buf][t * 32]))[i] = ((const uint4*)wG)[i];
#endif
  };

  v8f acc[4];
  #pragma unroll
  for (int jj = 0; jj < 4; ++jj) acc[jj] = v8f{0,0,0,0,0,0,0,0};

  stage(0, 0);
  wait_async0();
  __syncthreads();
  int cur = 0;
  for (int k0 = 0; k0 < 256; k0 += 32) {
    int nxt = cur ^ 1;
    if (k0 + 32 < 256) stage(k0 + 32, nxt);
    v16us a, bfr[4];
    #pragma unroll
    for (int i = 0; i < 16; ++i) a[i] = As[cur][(Mi * 16 + mrow) * 32 + kidxA(i, half)];
    #pragma unroll
    for (int jj = 0; jj < 4; ++jj) {
      int ncol = (nq * 4 + jj) * 16 + (lane & 15);
      #pragma unroll
      for (int i = 0; i < 16; ++i) bfr[jj][i] = Wt[cur][ncol * 32 + i + 16 * half];
    }
    #pragma unroll
    for (int jj = 0; jj < 4; ++jj) acc[jj] = wmma_bf16(a, bfr[jj], acc[jj]);
    wait_async0();
    __syncthreads();
    cur = nxt;
  }
  // epilogue: bias add, bounce through LDS (overlay on dead tile buffers)
  float* Ob = (float*)&Wt[0][0];                  // 32 KB overlay
  #pragma unroll
  for (int jj = 0; jj < 4; ++jj) {
    int ncol = (nq * 4 + jj) * 16 + (lane & 15);
    #pragma unroll
    for (int r = 0; r < 8; ++r)
      Ob[(Mi * 16 + 8 * half + r) * 256 + ncol] = acc[jj][r] + b_o[ncol];
  }
  __syncthreads();
  {
    const float4* srcv = (const float4*)(Ob + (t >> 3) * 256 + (t & 7) * 32);
    float4* dstv = (float4*)(out + (size_t)(mbase + (t >> 3)) * 256 + (t & 7) * 32);
    #pragma unroll
    for (int i = 0; i < 8; ++i) dstv[i] = srcv[i];
  }
}

// ---------------- launcher ----------------
extern "C" void kernel_launch(void* const* d_in, const int* in_sizes, int n_in,
                              void* d_out, int out_size, void* d_ws, size_t ws_size,
                              hipStream_t stream) {
  (void)in_sizes; (void)n_in; (void)out_size; (void)ws_size;
  const float* m      = (const float*)d_in[0];
  const float* z      = (const float*)d_in[1];
  const float* mask   = (const float*)d_in[2];
  const float* ln_m_g = (const float*)d_in[3];
  const float* ln_m_b = (const float*)d_in[4];
  const float* ln_z_g = (const float*)d_in[5];
  const float* ln_z_b = (const float*)d_in[6];
  const float* w_z    = (const float*)d_in[7];
  const float* w_q    = (const float*)d_in[8];
  const float* w_k    = (const float*)d_in[9];
  const float* w_v    = (const float*)d_in[10];
  const float* w_g    = (const float*)d_in[11];
  const float* b_g    = (const float*)d_in[12];
  const float* w_o    = (const float*)d_in[13];
  const float* b_o    = (const float*)d_in[14];
  float* out = (float*)d_out;

  char* ws = (char*)d_ws;
  unsigned short* mln  = (unsigned short*)(ws);              // 32 MB
  ws += (size_t)65536 * 256 * 2;
  unsigned short* qkvg = (unsigned short*)(ws);              // 128 MB
  ws += (size_t)65536 * 1024 * 2;
  unsigned short* og   = (unsigned short*)(ws);              // 32 MB
  ws += (size_t)65536 * 256 * 2;
  unsigned short* vT   = (unsigned short*)(ws);              // 32 MB [s][h][ch][key]
  ws += (size_t)2048 * 32 * 256 * 2;
  float* zb            = (float*)(ws);                       // 2 MB
  ws += (size_t)8 * 65536 * 4;
  unsigned short* wcT  = (unsigned short*)(ws);              // 512 KB [n][k]
  ws += (size_t)1024 * 256 * 2;
  unsigned short* woT  = (unsigned short*)(ws);              // 128 KB [n][k]
  ws += (size_t)256 * 256 * 2;

  k_ln_m <<<8192, 256, 0, stream>>>(m, ln_m_g, ln_m_b, mln);
  k_zbias<<<8192, 256, 0, stream>>>(z, ln_z_g, ln_z_b, w_z, zb);
  k_cvt_w<<<1280, 256, 0, stream>>>(w_q, w_k, w_v, w_g, w_o, wcT, woT);
  k_proj <<<dim3(2048, 4), 256, 0, stream>>>(mln, wcT, b_g, qkvg, vT);
  k_attn <<<2048, 256, 0, stream>>>(qkvg, vT, mask, zb, og);
  k_out  <<<2048, 256, 0, stream>>>(og, woT, b_o, out);
}
